// MetaConv2d_54322746359869
// MI455X (gfx1250) — matrix operand order repaired and checked by reading, not compiled, not run on hardware
//
#include <hip/hip_runtime.h>
#include <stdint.h>

typedef __attribute__((ext_vector_type(16))) __bf16 v16bf;
typedef __attribute__((ext_vector_type(8)))  float  v8f;

#define G_PAIRS 16
#define THREADS 512
#define S_DIM   64
#define CIN     64
#define COUT    64
#define KW      3
#define S_OUT   62
#define META    32
#define NCHUNK  6      // 192 / 32

// LDS layout (bytes)
#define XB_OFF   0                       // bf16 [16][64][64]  = 131072
#define WC_OFF   131072                  // bf16 [16][64][32]  =  65536
#define MF_OFF   196608                  // f32  [16][32]      =   2048
#define BI_OFF   198656                  // f32  [16][64]      =   4096
#define SMEM_BYTES 202752

// Pack two f32 -> bf16x2 (truncate) with one v_perm_b32:
// result = { hi16(f_hi), hi16(f_lo) }
__device__ __forceinline__ uint32_t pk2bf(float f_hi, float f_lo) {
  return __builtin_amdgcn_perm(__builtin_bit_cast(uint32_t, f_hi),
                               __builtin_bit_cast(uint32_t, f_lo),
                               0x07060302u);
}
__device__ __forceinline__ __bf16 bits2bf(uint16_t v) {
  return __builtin_bit_cast(__bf16, v);
}

// WMMA 16-bit fragment: 8 dwords = 16 bf16 per lane
union Frag {
  uint4    q[2];
  uint32_t u[8];
  v16bf    v;
};

__global__ __launch_bounds__(THREADS)
void metaconv_fused_kernel(const float* __restrict__ meta,
                           const float* __restrict__ x,
                           const float* __restrict__ wlw,
                           const float* __restrict__ wlb,
                           const float* __restrict__ blw,
                           const float* __restrict__ blb,
                           float* __restrict__ out) {
  extern __shared__ char smem[];
  __bf16* x_b  = (__bf16*)(smem + XB_OFF);
  __bf16* w_c  = (__bf16*)(smem + WC_OFF);
  float*  m_f  = (float*)(smem + MF_OFF);
  float*  bi_f = (float*)(smem + BI_OFF);

  const int tid   = threadIdx.x;
  const int wave  = tid >> 5;     // 0..15, wave g owns pair g for the conv
  const int lane  = tid & 31;
  const int pair0 = blockIdx.x * G_PAIRS;

  // ---- Phase 0: meta -> LDS (f32, [16][32]; 512 floats, 1/thread) ----
  m_f[tid] = meta[(size_t)pair0 * META + tid];
  __syncthreads();

  // ---- Phase 0b: stage x as bf16 [g][s][c]; and bias[g][o] ----
  {
    const float2* xg = (const float2*)(x + (size_t)pair0 * (S_DIM * CIN));
    uint32_t* xb32 = (uint32_t*)x_b;
#pragma unroll 4
    for (int i = 0; i < 64; ++i) {
      int grp = tid + i * THREADS;           // 32768 float2 groups
      float2 v = xg[grp];
      xb32[grp] = pk2bf(v.y, v.x);
    }
    for (int v = tid; v < G_PAIRS * COUT; v += THREADS) {
      int g = v >> 6, o = v & 63;
      float s = blb[o];
#pragma unroll
      for (int m = 0; m < META; ++m) s += m_f[g * META + m] * blw[o * META + m];
      bi_f[v] = s;
    }
  }
  __syncthreads();

  // ---- Hypernet A fragment (16 pairs x 32 meta, bf16), identical per wave ----
  // 16-bit A 16x32 layout: lanes 0-15 M=lane, v0-3:K=0..7, v4-7:K=16..23;
  // lanes 16-31 shift K by +8.
  Frag a_h;
  {
    const int M = lane & 15;
    const int koff = (lane >= 16) ? 8 : 0;
#pragma unroll
    for (int j = 0; j < 8; ++j) {
      int k0 = ((j < 4) ? 2 * j : 16 + 2 * (j - 4)) + koff;
      a_h.u[j] = pk2bf(m_f[M * META + k0 + 1], m_f[M * META + k0]);
    }
  }

  v8f acc[4][4];
#pragma unroll
  for (int mt = 0; mt < 4; ++mt)
#pragma unroll
    for (int nt = 0; nt < 4; ++nt) acc[mt][nt] = (v8f){};

  // ck ordering: ck = k*64 + c  (chunk ch: k = ch/2 fixed, c base = (ch&1)*32)
  for (int ch = 0; ch < NCHUNK; ++ch) {
    const int kk = ch >> 1;
    const int cb = (ch & 1) * 32;

    // ---- Hypernet: compute w chunk [g][o][ckLocal] via WMMA; 8 tiles/wave ----
#pragma unroll
    for (int q = 0; q < 8; ++q) {
      const int tI  = wave * 8 + q;          // 0..127
      const int o   = tI >> 1;
      const int n0  = (tI & 1) * 16;
      const int col = lane & 15;
      const int mb  = (lane >= 16) ? 8 : 0;
      const int cL  = n0 + col;              // ckLocal 0..31
      const int p   = o * (CIN * KW) + (cb + cL) * KW + kk;
      const float4* w4 = (const float4*)(wlw + (size_t)p * META + mb);
      __builtin_prefetch(wlw + ((size_t)p + 96) * META, 0, 1);
      float4 q0 = w4[0], q1 = w4[1], q2 = w4[4], q3 = w4[5];
      Frag b_h;
      b_h.u[0] = pk2bf(q0.y, q0.x);
      b_h.u[1] = pk2bf(q0.w, q0.z);
      b_h.u[2] = pk2bf(q1.y, q1.x);
      b_h.u[3] = pk2bf(q1.w, q1.z);
      b_h.u[4] = pk2bf(q2.y, q2.x);
      b_h.u[5] = pk2bf(q2.w, q2.z);
      b_h.u[6] = pk2bf(q3.y, q3.x);
      b_h.u[7] = pk2bf(q3.w, q3.z);
      v8f c0 = (v8f){};
      v8f d = __builtin_amdgcn_wmma_f32_16x16x32_bf16(
          false, a_h.v, false, b_h.v, (short)0, c0, false, false);
      const float wb = wlb[p];
      const int gg = (lane >= 16) ? 8 : 0;
#pragma unroll
      for (int r = 0; r < 8; ++r) {
        int g = r + gg;   // D tile rows = pair index
        uint32_t bits = __builtin_bit_cast(uint32_t, d[r] + wb);
        w_c[((g * COUT + o) << 5) + cL] = bits2bf((uint16_t)(bits >> 16));
      }
    }
    __syncthreads();

    // ---- Conv: wave g consumes its pair; 4x4 WMMA tiles, K-step = this chunk ----
    {
      const int g = wave;
      Frag bfr[4];
#pragma unroll
      for (int nt = 0; nt < 4; ++nt) {
        const int oc  = nt * 16 + (lane & 15);
        const int ckb = (lane >= 16) ? 4 : 0;  // dword offset (ck +8)
        const uint32_t* wrow32 = (const uint32_t*)(w_c + ((g * COUT + oc) << 5));
        bfr[nt].q[0] = *(const uint4*)(wrow32 + ckb);
        bfr[nt].q[1] = *(const uint4*)(wrow32 + ckb + 8);
      }
      Frag afr[4];
#pragma unroll
      for (int mt = 0; mt < 4; ++mt) {
        const int t = mt * 16 + (lane & 15);
        const int kq = (lane >= 16) ? 4 : 0;   // dword offset (ck +8)
        int s = t + kk; if (s > 63) s = 63;    // rows t>=62 are discarded anyway
        const uint32_t* xrow32 =
            (const uint32_t*)(x_b + ((g * S_DIM + s) << 6) + cb);
        afr[mt].q[0] = *(const uint4*)(xrow32 + kq);       // ck 0..7 (+8)
        afr[mt].q[1] = *(const uint4*)(xrow32 + 8 + kq);   // ck 16..23 (+8)
      }
#pragma unroll
      for (int mt = 0; mt < 4; ++mt)
#pragma unroll
        for (int nt = 0; nt < 4; ++nt)
          acc[mt][nt] = __builtin_amdgcn_wmma_f32_16x16x32_bf16(
              false, afr[mt].v, false, bfr[nt].v, (short)0, acc[mt][nt],
              false, false);
    }
    __syncthreads();
  }

  // ---- Epilogue: add bias, store valid rows t < 62 ----
  {
    const int g = wave;
    const size_t base = ((size_t)(pair0 + g)) * S_OUT * COUT;
    const int rb = (lane >= 16) ? 8 : 0;
#pragma unroll
    for (int nt = 0; nt < 4; ++nt) {
      const int o = nt * 16 + (lane & 15);
      const float bb = bi_f[g * COUT + o];
#pragma unroll
      for (int mt = 0; mt < 4; ++mt) {
#pragma unroll
        for (int r = 0; r < 8; ++r) {
          int t = mt * 16 + rb + r;
          if (t < S_OUT) out[base + (size_t)t * COUT + o] = acc[mt][nt][r] + bb;
        }
      }
    }
  }
}

extern "C" void kernel_launch(void* const* d_in, const int* in_sizes, int n_in,
                              void* d_out, int out_size, void* d_ws, size_t ws_size,
                              hipStream_t stream) {
  const float* meta = (const float*)d_in[0];
  const float* x    = (const float*)d_in[1];
  const float* wlw  = (const float*)d_in[2];
  const float* wlb  = (const float*)d_in[3];
  const float* blw  = (const float*)d_in[4];
  const float* blb  = (const float*)d_in[5];
  float* out = (float*)d_out;

  (void)in_sizes; (void)n_in; (void)out_size; (void)d_ws; (void)ws_size;

  hipFuncSetAttribute((const void*)metaconv_fused_kernel,
                      hipFuncAttributeMaxDynamicSharedMemorySize, SMEM_BYTES);

  const int n_pairs = 32 * 256;                 // B * N
  dim3 grid(n_pairs / G_PAIRS);                 // 512 workgroups
  dim3 block(THREADS);                          // 16 waves (wave32)
  metaconv_fused_kernel<<<grid, block, SMEM_BYTES, stream>>>(
      meta, x, wlw, wlb, blw, blb, out);
}
